// PoolingFunction_12962211299760
// MI455X (gfx1250) — matrix operand order repaired and checked by reading, not compiled, not run on hardware
//
#include <hip/hip_runtime.h>

// ---------------- problem constants ----------------
#define B_SZ   8
#define S_LEN  4096
#define T_LEN  512
#define HIDD   512
#define HEADS  8
#define DKH    64

typedef __attribute__((ext_vector_type(16))) __bf16 v16bf;
typedef __attribute__((ext_vector_type(8)))  float  v8f;

union AB {            // one WMMA A/B operand: 16 bf16 = 32 bytes = 8 VGPRs
    v16bf          v;
    unsigned short u[16];
    uint4          q4[2];
};
union F8 { float4 q[2]; float f[8]; };

// float -> bf16, round-to-nearest-even (bit math; no __bf16 arithmetic)
__device__ __forceinline__ unsigned short f2bf(float f) {
    unsigned int x = __float_as_uint(f);
    x += 0x7FFFu + ((x >> 16) & 1u);
    return (unsigned short)(x >> 16);
}

__device__ __forceinline__ v8f wmma_bf16(const AB& a, const AB& b, v8f c) {
    return __builtin_amdgcn_wmma_f32_16x16x32_bf16(
        false, a.v, false, b.v, (short)0, c, false, false);
}

// cross-lane xor shuffle within 16-lane groups via ds_swizzle
template <int IMM>
__device__ __forceinline__ float swzf(float v) {
    return __int_as_float(__builtin_amdgcn_ds_swizzle(__float_as_int(v), IMM));
}
__device__ __forceinline__ float redmax16(float v) {
    v = fmaxf(v, swzf<0x041F>(v));
    v = fmaxf(v, swzf<0x081F>(v));
    v = fmaxf(v, swzf<0x101F>(v));
    v = fmaxf(v, swzf<0x201F>(v));
    return v;
}
__device__ __forceinline__ float redsum16(float v) {
    v += swzf<0x041F>(v);
    v += swzf<0x081F>(v);
    v += swzf<0x101F>(v);
    v += swzf<0x201F>(v);
    return v;
}

// CDNA5 async global->LDS copy of 16 bytes (tracked by ASYNCcnt).
// LDS dest offset = low 32 bits of the generic pointer (LDS aperture rule).
__device__ __forceinline__ void async_cp16(unsigned short* ldsdst,
                                           const unsigned short* gsrc) {
    unsigned lo = (unsigned)(unsigned long long)(uintptr_t)ldsdst;
    unsigned long long ga = (unsigned long long)(uintptr_t)gsrc;
    asm volatile("global_load_async_to_lds_b128 %0, %1, off"
                 :: "v"(lo), "v"(ga) : "memory");
}
__device__ __forceinline__ void wait_async0() {
    asm volatile("s_wait_asynccnt 0" ::: "memory");
}

// ------------------------------------------------------------------
// elementwise f32 -> bf16 (8 elements / thread, fully vectorized)
// ------------------------------------------------------------------
__global__ void cvt_f32_to_bf16(const float* __restrict__ x,
                                unsigned short* __restrict__ y, int n8) {
    const int i = blockIdx.x * blockDim.x + threadIdx.x;
    if (i >= n8) return;
    const float* xp = x + (size_t)i * 8;
    F8 v; v.q[0] = *(const float4*)xp; v.q[1] = *(const float4*)(xp + 4);
    union { unsigned short u[8]; uint4 q; } o;
#pragma unroll
    for (int k = 0; k < 8; ++k) o.u[k] = f2bf(v.f[k]);
    *(uint4*)(y + (size_t)i * 8) = o.q;
}

// ------------------------------------------------------------------
// all-bf16 GEMM: Y[m,n] = sum_k X[m,k]*W[n,k] + bias[n], K = N = 512
// OUTMODE: 0 = bf16 [M,512]; 1 = bf16 transposed [512,M]; 2 = f32 [M,512]
// block 256 thr (8 waves) -> 64x32 tile; one 16x16 WMMA tile per wave
// ------------------------------------------------------------------
template <int OUTMODE>
__global__ void gemm512_wmma(const unsigned short* __restrict__ X,
                             const unsigned short* __restrict__ Wb,
                             const float* __restrict__ bias,
                             void* __restrict__ Yv, int M) {
    const int lane = threadIdx.x & 31;
    const int wid  = threadIdx.x >> 5;
    const int l16  = lane & 15;
    const int half = lane >> 4;
    const int mBase = blockIdx.x * 64 + (wid & 3) * 16;
    const int nBase = blockIdx.y * 32 + (wid >> 2) * 16;
    const int m = mBase + l16;
    const int n = nBase + l16;

    v8f acc = {0.f, 0.f, 0.f, 0.f, 0.f, 0.f, 0.f, 0.f};
    const unsigned short* xr = X  + (size_t)m * 512;
    const unsigned short* wr = Wb + (size_t)n * 512;

#pragma unroll 4
    for (int k0 = 0; k0 < 512; k0 += 32) {
        AB a, b;
        // A: k = k0 + half*8 + (i&7) + (i>>3)*16
        a.q4[0] = *(const uint4*)(xr + k0 + half * 8);
        a.q4[1] = *(const uint4*)(xr + k0 + half * 8 + 16);
        // B: k = k0 + half*8 + (i&7) + (i>>3)*16
        b.q4[0] = *(const uint4*)(wr + k0 + half * 8);
        b.q4[1] = *(const uint4*)(wr + k0 + half * 8 + 16);
        acc = wmma_bf16(a, b, acc);
    }

    const float bb = bias[n];
#pragma unroll
    for (int r = 0; r < 8; ++r) {
        const float val = acc[r] + bb;
        const int mr = mBase + r + half * 8;
        if constexpr (OUTMODE == 2) {
            ((float*)Yv)[(size_t)mr * 512 + n] = val;
        } else if constexpr (OUTMODE == 1) {
            ((unsigned short*)Yv)[(size_t)n * M + mr] = f2bf(val);
        } else {
            ((unsigned short*)Yv)[(size_t)mr * 512 + n] = f2bf(val);
        }
    }
}

// ------------------------------------------------------------------
// Flash attention. Block = 4 waves sharing one (b,h); waves take 4
// consecutive 16-row t-tiles. 64-column K/V chunks are staged into LDS
// once per block via async global->LDS copies, double-buffered so the
// copy of chunk n+1 overlaps WMMA work on chunk n. Row-sum reduction
// is deferred (per-lane partials) to keep the hot loop WMMA-dominated.
// grid = B*H*(T/64) = 512 blocks, 128 threads.
// ------------------------------------------------------------------
__global__ void attn_flash_wmma(const unsigned short* __restrict__ Qb,
                                const unsigned short* __restrict__ Kb,
                                const unsigned short* __restrict__ Vt,
                                const unsigned char* __restrict__ mask,
                                unsigned short* __restrict__ ctx) {
    __shared__ __align__(16) unsigned short Ks[2][64 * DKH];  // [s][d] 8KB ea
    __shared__ __align__(16) unsigned short Vs[2][DKH * 64];  // [d][s] 8KB ea
    __shared__ __align__(16) unsigned short Pl[4][16 * 64];   // per-wave P

    const int tid  = threadIdx.x;
    const int lane = tid & 31;
    const int wid  = tid >> 5;
    const int l16  = lane & 15;
    const int half = lane >> 4;

    const int blocksPerBH = T_LEN / 64;            // 8
    const int bh = blockIdx.x / blocksPerBH;
    const int b  = bh / HEADS;
    const int h  = bh % HEADS;
    const int t0 = (blockIdx.x % blocksPerBH) * 64 + wid * 16;

    const size_t MKV = (size_t)B_SZ * S_LEN;
    const unsigned short* kbp = Kb + (size_t)b * S_LEN * HIDD + h * DKH;
    const unsigned short* vbp = Vt + (size_t)(h * DKH) * MKV + (size_t)b * S_LEN;

    // ---- Q fragments (16 x 64 -> two K=32 A-fragments), kept in regs ----
    AB aq[2];
    {
        const unsigned short* qr =
            Qb + (size_t)(b * T_LEN + t0 + l16) * HIDD + h * DKH;
#pragma unroll
        for (int j = 0; j < 2; ++j) {
            const int ka = j * 32 + half * 8;
            aq[j].q4[0] = *(const uint4*)(qr + ka);
            aq[j].q4[1] = *(const uint4*)(qr + ka + 16);
        }
    }

    float rmax[8], rsum[8];           // rsum is a PER-LANE partial
    v8f   acc[4];
#pragma unroll
    for (int r = 0; r < 8; ++r) { rmax[r] = -1e30f; rsum[r] = 0.f; }
#pragma unroll
    for (int dt = 0; dt < 4; ++dt) acc[dt] = (v8f){0.f,0.f,0.f,0.f,0.f,0.f,0.f,0.f};

    // cooperative async staging of one 64-column K/V chunk (16 KB)
    auto stage = [&](int buf, int s0) {
#pragma unroll
        for (int t = 0; t < 4; ++t) {            // K: 64 x 64 halves = 512 x 16B
            const int i = tid + t * 128;
            const int row = i >> 3, c = (i & 7) * 8;
            async_cp16(&Ks[buf][row * DKH + c],
                       kbp + (size_t)(s0 + row) * HIDD + c);
        }
#pragma unroll
        for (int t = 0; t < 4; ++t) {            // V: 64 x 64 halves = 512 x 16B
            const int i = tid + t * 128;
            const int row = i >> 3, c = (i & 7) * 8;
            async_cp16(&Vs[buf][row * 64 + c],
                       vbp + (size_t)row * MKV + s0 + c);
        }
    };

    stage(0, 0);
    wait_async0();
    __syncthreads();

    const float SC = 0.125f;  // 1/sqrt(64)

    for (int s0 = 0; s0 < S_LEN; s0 += 64) {
        const int cur = (s0 >> 6) & 1;
        if (s0 + 64 < S_LEN) stage(cur ^ 1, s0 + 64);   // overlap copy w/ compute

        // ---------- scores: four 16x16 C tiles over K=64, from LDS ----------
        v8f cs[4];
#pragma unroll
        for (int st = 0; st < 4; ++st) {
            cs[st] = (v8f){0.f,0.f,0.f,0.f,0.f,0.f,0.f,0.f};
#pragma unroll
            for (int j = 0; j < 2; ++j) {
                AB bk;
                const unsigned short* kr =
                    &Ks[cur][(st * 16 + l16) * DKH + j * 32 + half * 8];
                bk.q4[0] = *(const uint4*)kr; bk.q4[1] = *(const uint4*)(kr + 16);
                cs[st] = wmma_bf16(aq[j], bk, cs[st]);
            }
        }

        float mb[4];
#pragma unroll
        for (int st = 0; st < 4; ++st)
            mb[st] = mask[(size_t)b * S_LEN + s0 + st * 16 + l16] ? 0.f : -1e20f;

        // ---------- online softmax (row-max reduced; row-sum per-lane) ----------
#pragma unroll
        for (int r = 0; r < 8; ++r) {
            cs[0][r] = cs[0][r] * SC + mb[0];
            cs[1][r] = cs[1][r] * SC + mb[1];
            cs[2][r] = cs[2][r] * SC + mb[2];
            cs[3][r] = cs[3][r] * SC + mb[3];
            const float cm = redmax16(fmaxf(fmaxf(cs[0][r], cs[1][r]),
                                            fmaxf(cs[2][r], cs[3][r])));
            const float mnew = fmaxf(rmax[r], cm);
            const float corr = __expf(rmax[r] - mnew);
            rmax[r] = mnew;
            cs[0][r] = __expf(cs[0][r] - mnew);
            cs[1][r] = __expf(cs[1][r] - mnew);
            cs[2][r] = __expf(cs[2][r] - mnew);
            cs[3][r] = __expf(cs[3][r] - mnew);
            rsum[r] = rsum[r] * corr +
                      (cs[0][r] + cs[1][r]) + (cs[2][r] + cs[3][r]);
            acc[0][r] *= corr; acc[1][r] *= corr;
            acc[2][r] *= corr; acc[3][r] *= corr;
        }

        // ---------- stage P (16 x 64 bf16) in this wave's LDS region ----------
        unsigned short* pt = &Pl[wid][0];
#pragma unroll
        for (int r = 0; r < 8; ++r) {
            const int row = r + half * 8;
#pragma unroll
            for (int st = 0; st < 4; ++st)
                pt[row * 64 + st * 16 + l16] = f2bf(cs[st][r]);
        }
        AB ap[2];  // same-wave LDS ordering is in-order via DScnt
#pragma unroll
        for (int j = 0; j < 2; ++j) {
            const unsigned short* pr = pt + l16 * 64 + j * 32 + half * 8;
            ap[j].q4[0] = *(const uint4*)pr;
            ap[j].q4[1] = *(const uint4*)(pr + 16);
        }

        // ---------- ctx += P @ V (K = 64 -> 2 WMMAs per d-tile) ----------
#pragma unroll
        for (int dt = 0; dt < 4; ++dt) {
#pragma unroll
            for (int j = 0; j < 2; ++j) {
                AB bvf;
                const unsigned short* vr =
                    &Vs[cur][(dt * 16 + l16) * 64 + j * 32 + half * 8];
                bvf.q4[0] = *(const uint4*)vr; bvf.q4[1] = *(const uint4*)(vr + 16);
                acc[dt] = wmma_bf16(ap[j], bvf, acc[dt]);
            }
        }

        wait_async0();      // next buffer fully landed in LDS
        __syncthreads();    // everyone done reading `cur` before it is reused
    }

    // ---------- final lane-reduction of row sums, normalize, store ----------
#pragma unroll
    for (int r = 0; r < 8; ++r) {
        const float inv = 1.0f / redsum16(rsum[r]);
        const size_t row = (size_t)(b * T_LEN + t0 + r + half * 8);
#pragma unroll
        for (int dt = 0; dt < 4; ++dt) {
            ctx[row * HIDD + h * DKH + dt * 16 + l16] = f2bf(acc[dt][r] * inv);
        }
    }
}

// ------------------------------------------------------------------
extern "C" void kernel_launch(void* const* d_in, const int* in_sizes, int n_in,
                              void* d_out, int out_size, void* d_ws, size_t ws_size,
                              hipStream_t stream) {
    (void)in_sizes; (void)n_in; (void)out_size; (void)ws_size;
    const float*         inputs  = (const float*)d_in[0];        // B x S x 512
    const float*         targets = (const float*)d_in[1];        // B x T x 512
    const unsigned char* mask    = (const unsigned char*)d_in[2];
    const float* Wq = (const float*)d_in[3];
    const float* bq = (const float*)d_in[4];
    const float* Wk = (const float*)d_in[5];
    const float* bk = (const float*)d_in[6];
    const float* Wv = (const float*)d_in[7];
    const float* bv = (const float*)d_in[8];
    const float* Wo = (const float*)d_in[9];
    const float* bo = (const float*)d_in[10];
    float* out = (float*)d_out;

    const int MQ  = B_SZ * T_LEN;   // 4096
    const int MKV = B_SZ * S_LEN;   // 32768

    // workspace layout (bf16 halves)
    unsigned short* p   = (unsigned short*)d_ws;
    unsigned short* Xb  = p;  p += (size_t)MKV * HIDD;   // inputs  bf16
    unsigned short* Tb  = p;  p += (size_t)MQ  * HIDD;   // targets bf16
    unsigned short* Wqb = p;  p += (size_t)HIDD * HIDD;
    unsigned short* Wkb = p;  p += (size_t)HIDD * HIDD;
    unsigned short* Wvb = p;  p += (size_t)HIDD * HIDD;
    unsigned short* Wob = p;  p += (size_t)HIDD * HIDD;
    unsigned short* Qf  = p;  p += (size_t)MQ  * HIDD;   // Q bf16 [MQ,512]
    unsigned short* Kf  = p;  p += (size_t)MKV * HIDD;   // K bf16 [MKV,512]
    unsigned short* Vtf = p;  p += (size_t)HIDD * MKV;   // V bf16 transposed [512,MKV]
    unsigned short* ctx = p;                             // ctx bf16 [MQ,512]

    // ---- 1) one-time f32 -> bf16 conversions ----
    {
        const int thr = 256;
        int n8;
        n8 = MKV * HIDD / 8;
        cvt_f32_to_bf16<<<dim3((n8 + thr - 1) / thr), dim3(thr), 0, stream>>>(inputs, Xb, n8);
        n8 = MQ * HIDD / 8;
        cvt_f32_to_bf16<<<dim3((n8 + thr - 1) / thr), dim3(thr), 0, stream>>>(targets, Tb, n8);
        n8 = HIDD * HIDD / 8;
        cvt_f32_to_bf16<<<dim3((n8 + thr - 1) / thr), dim3(thr), 0, stream>>>(Wq, Wqb, n8);
        cvt_f32_to_bf16<<<dim3((n8 + thr - 1) / thr), dim3(thr), 0, stream>>>(Wk, Wkb, n8);
        cvt_f32_to_bf16<<<dim3((n8 + thr - 1) / thr), dim3(thr), 0, stream>>>(Wv, Wvb, n8);
        cvt_f32_to_bf16<<<dim3((n8 + thr - 1) / thr), dim3(thr), 0, stream>>>(Wo, Wob, n8);
    }

    // ---- 2) projections (all-bf16 WMMA GEMMs) ----
    dim3 blk(256);
    gemm512_wmma<0><<<dim3(MQ / 64, 512 / 32),  blk, 0, stream>>>(Tb, Wqb, bq, Qf, MQ);
    gemm512_wmma<0><<<dim3(MKV / 64, 512 / 32), blk, 0, stream>>>(Xb, Wkb, bk, Kf, MKV);
    gemm512_wmma<1><<<dim3(MKV / 64, 512 / 32), blk, 0, stream>>>(Xb, Wvb, bv, Vtf, MKV);

    // ---- 3) flash attention (async-LDS staged, double buffered) ----
    attn_flash_wmma<<<dim3(B_SZ * HEADS * (T_LEN / 64)), dim3(128), 0, stream>>>(
        Qf, Kf, Vtf, mask, ctx);

    // ---- 4) output projection ----
    gemm512_wmma<2><<<dim3(MQ / 64, 512 / 32), blk, 0, stream>>>(ctx, Wob, bo, out, MQ);
}
